// FuzzyTM_13022340842129
// MI455X (gfx1250) — compile-verified, matrix operand order checked by reading, not run on hardware
//
#include <hip/hip_runtime.h>
#include <hip/hip_bf16.h>

// FuzzyTM forward scan for MI455X (gfx1250, wave32).
//
// Roofline: 137 GFLOP total vs 128 GB of table reads per full scan (bf16).
// Both softmaxed transition tables (2 x 4096x4096 bf16 = 64 MB) fit in the
// 192 MB L2, so the 4096-step sequential scan streams entirely from L2;
// HBM only sees the one-time softmax pass. WMMA (bf16 16x16x32) does the
// per-step matvec with the state broadcast into all 16 A rows.

#define S      4096
#define SEQ    4096
#define BLOCKS 64     // persistent grid; 64 blocks are always co-resident
#define TPB    128    // 4 wave32s; each wave owns 16 output columns

typedef __attribute__((ext_vector_type(16))) __bf16 v16bf;
typedef __attribute__((ext_vector_type(8)))  __bf16 v8bf;
typedef __attribute__((ext_vector_type(8)))  float  v8f;

// ---------------------------------------------------------------------------
// Kernel 1: softmax each contiguous row T[s, sym, :] (4096 f32) and store the
// result TRANSPOSED in bf16: Wt[sym][j*S + s].  Transposed layout makes every
// WMMA B-operand lane a contiguous 16-byte load in the scan kernel.
// ---------------------------------------------------------------------------
__global__ void softmaxT_kernel(const float* __restrict__ T,
                                __bf16* __restrict__ Wt) {
  const int row = blockIdx.x;            // row = s*2 + sym
  const int s   = row >> 1;
  const int sym = row & 1;
  const float* __restrict__ x = T + (size_t)row * S;
  __shared__ float red[256];
  const int tid = threadIdx.x;

  float m = -3.0e38f;
  for (int j = tid; j < S; j += 256) m = fmaxf(m, x[j]);
  red[tid] = m; __syncthreads();
#pragma unroll
  for (int w = 128; w > 0; w >>= 1) {
    if (tid < w) red[tid] = fmaxf(red[tid], red[tid + w]);
    __syncthreads();
  }
  m = red[0];
  __syncthreads();

  float sum = 0.0f;
  for (int j = tid; j < S; j += 256) sum += __expf(x[j] - m);
  red[tid] = sum; __syncthreads();
#pragma unroll
  for (int w = 128; w > 0; w >>= 1) {
    if (tid < w) red[tid] += red[tid + w];
    __syncthreads();
  }
  const float inv = 1.0f / red[0];

  __bf16* __restrict__ Wsym = Wt + (size_t)sym * S * S;
  for (int j = tid; j < S; j += 256)
    Wsym[(size_t)j * S + s] = (__bf16)(__expf(x[j] - m) * inv);
}

// ---------------------------------------------------------------------------
// Kernel 2: softmax O rows (4 wide), build one-hot initial state (f32),
// zero the grid-sync counter (re-run every launch -> deterministic replays).
// ---------------------------------------------------------------------------
__global__ void softmaxO_init_kernel(const float* __restrict__ O,
                                     float* __restrict__ Om,
                                     float* __restrict__ st0,
                                     unsigned* __restrict__ sync,
                                     const int* __restrict__ init_idx) {
  const int g = blockIdx.x * blockDim.x + threadIdx.x;
  if (g < 2 * S) {
    const float* x = O + (size_t)g * 4;
    float a = x[0], b = x[1], c = x[2], d = x[3];
    float m  = fmaxf(fmaxf(a, b), fmaxf(c, d));
    float e0 = __expf(a - m), e1 = __expf(b - m);
    float e2 = __expf(c - m), e3 = __expf(d - m);
    float inv = 1.0f / (e0 + e1 + e2 + e3);
    float* y = Om + (size_t)g * 4;
    y[0] = e0 * inv; y[1] = e1 * inv; y[2] = e2 * inv; y[3] = e3 * inv;
  }
  if (g < S) st0[g] = (g == *init_idx) ? 1.0f : 0.0f;
  if (g == 0) *sync = 0u;
}

// ---------------------------------------------------------------------------
// Kernel 3: persistent sequential scan.
//   Per step t:  new_state[j] = sum_s state[s] * Wt[sym][j][s]
//   Each wave computes 16 columns via v_wmma_f32_16x16x32_bf16 with the state
//   chunk broadcast across all 16 A rows (redundant FLOPs are free: the step
//   is L2-bandwidth bound).  Grid barrier = monotonic atomic counter.
// A-layout per ISA 7.12.2 (16-bit A 16x32): lanes 0-15 hold K=8h+[0..7] in
// VGPRs 0-3 and K=16+8h+[0..7] in VGPRs 4-7 (h = lane>>4); B mirrors with
// N = lane&15 and K pairs packed per dword -> two 16B loads per lane each.
// ---------------------------------------------------------------------------
__global__ void scan_kernel(const __bf16* __restrict__ Wt,
                            const float*  __restrict__ Om,
                            const int*    __restrict__ seq,
                            float* __restrict__ stateA,
                            float* __restrict__ stateB,
                            float* __restrict__ out,
                            unsigned* __restrict__ sync) {
  __shared__ __bf16 s_state[S];
  __shared__ float  s_red[4];

  const int tid  = threadIdx.x;
  const int lane = tid & 31;
  const int wave = tid >> 5;
  const int n    = lane & 15;          // B column / D column within tile
  const int h    = lane >> 4;          // lane half selects K sub-block
  const int j0   = blockIdx.x * 64 + wave * 16;  // this wave's 16 columns
  const int off1 = 8 * h;              // elements (bf16) -> 16B aligned
  const int off2 = 16 + 8 * h;

  float* cur = stateA;
  float* nxt = stateB;
  unsigned target = 0;

  for (int t = 0; t < SEQ; ++t) {
    const int sym = seq[t];

    // Stage current state (f32, global) into LDS as bf16 for the A operand.
    for (int i = tid; i < S; i += TPB) s_state[i] = (__bf16)cur[i];
    __syncthreads();

    // Block 0 additionally emits out_prob[t] = state @ Om[:,sym,:]  (4 vals).
    if (blockIdx.x == 0) {
      float p0 = 0.f, p1 = 0.f, p2 = 0.f, p3 = 0.f;
      for (int s2 = tid; s2 < S; s2 += TPB) {
        const float st = cur[s2];
        const float* om = Om + (((size_t)s2 * 2 + sym) * 4);
        p0 += st * om[0]; p1 += st * om[1];
        p2 += st * om[2]; p3 += st * om[3];
      }
      if (tid < 4) s_red[tid] = 0.0f;
      __syncthreads();
      atomicAdd(&s_red[0], p0);        // ds_add_f32
      atomicAdd(&s_red[1], p1);
      atomicAdd(&s_red[2], p2);
      atomicAdd(&s_red[3], p3);
      __syncthreads();
      if (tid < 4) out[(size_t)t * 4 + tid] = s_red[tid];
    }

    // ---- 16-column matvec via WMMA, K = 4096 in steps of 32 --------------
    const __bf16* __restrict__ brow =
        Wt + (size_t)sym * S * S + (size_t)(j0 + n) * S;  // column j0+n, contiguous in s

    v8f acc = {};
    v8bf a_lo = *(const v8bf*)(s_state + off1);
    v8bf a_hi = *(const v8bf*)(s_state + off2);
    v8bf b_lo = *(const v8bf*)(brow + off1);
    v8bf b_hi = *(const v8bf*)(brow + off2);

    for (int k0 = 0; k0 < S; k0 += 32) {
      v16bf a = __builtin_shufflevector(a_lo, a_hi,
                  0,1,2,3,4,5,6,7,8,9,10,11,12,13,14,15);
      v16bf b = __builtin_shufflevector(b_lo, b_hi,
                  0,1,2,3,4,5,6,7,8,9,10,11,12,13,14,15);
      const int kn = k0 + 32;
      if (kn < S) {                    // software-pipelined next-tile loads
        a_lo = *(const v8bf*)(s_state + kn + off1);
        a_hi = *(const v8bf*)(s_state + kn + off2);
        b_lo = *(const v8bf*)(brow + kn + off1);
        b_hi = *(const v8bf*)(brow + kn + off2);
      }
      acc = __builtin_amdgcn_wmma_f32_16x16x32_bf16(
                false, a, false, b, (short)0, acc, false, false);
    }

    // D row M=0: VGPR0 of lanes 0-15, N = lane.  All 16 rows are identical.
    if (lane < 16) nxt[j0 + lane] = acc[0];

    __syncthreads();                   // all waves done with s_state
    // ---- grid-wide barrier (monotonic counter, s_sleep backoff) ----------
    target += gridDim.x;
    if (tid == 0) {
      __threadfence();
      __hip_atomic_fetch_add(sync, 1u, __ATOMIC_ACQ_REL,
                             __HIP_MEMORY_SCOPE_AGENT);
      while (__hip_atomic_load(sync, __ATOMIC_ACQUIRE,
                               __HIP_MEMORY_SCOPE_AGENT) < target) {
        __builtin_amdgcn_s_sleep(8);
      }
    }
    __syncthreads();

    float* tmp = cur; cur = nxt; nxt = tmp;
  }
}

// ---------------------------------------------------------------------------
extern "C" void kernel_launch(void* const* d_in, const int* in_sizes, int n_in,
                              void* d_out, int out_size, void* d_ws, size_t ws_size,
                              hipStream_t stream) {
  const float* T        = (const float*)d_in[0];   // (S, 2, S) f32
  const float* O        = (const float*)d_in[1];   // (S, 2, 4) f32
  const int*   init_idx = (const int*)  d_in[2];   // scalar
  const int*   seq      = (const int*)  d_in[3];   // (SEQ,) i32
  float*       out      = (float*)d_out;           // (SEQ, 4) f32

  char* ws = (char*)d_ws;
  size_t off = 0;
  __bf16* Wt = (__bf16*)(ws + off); off += (size_t)2 * S * S * sizeof(__bf16); // 64 MB
  float*  Om = (float*) (ws + off); off += (size_t)S * 2 * 4 * sizeof(float);
  float*  stA = (float*)(ws + off); off += (size_t)S * sizeof(float);
  float*  stB = (float*)(ws + off); off += (size_t)S * sizeof(float);
  off = (off + 127) & ~(size_t)127;
  unsigned* sync = (unsigned*)(ws + off);

  softmaxT_kernel<<<2 * S, 256, 0, stream>>>(T, Wt);
  softmaxO_init_kernel<<<(2 * S) / 256, 256, 0, stream>>>(O, Om, stA, sync, init_idx);
  scan_kernel<<<BLOCKS, TPB, 0, stream>>>(Wt, Om, seq, stA, stB, out, sync);
}
